// SemanticClipAttnProcessor_5617817223511
// MI455X (gfx1250) — matrix operand order repaired and checked by reading, not compiled, not run on hardware
//
#include <hip/hip_runtime.h>
#include <hip/hip_bf16.h>
#include <math.h>

// ---------------- problem constants (from reference) ----------------
#define BB      2
#define SS      1024
#define CC      1280
#define T81     81
#define TT      77      // text tokens
#define TTP     160     // B*TT = 154 padded up to multiple of 32
#define NTOK    4       // ip tokens
#define EE      2048
#define HH      20
#define DH      64      // CC / HH
#define SEM_SCALE 0.5f
#define IP_SCALE  1.0f
#define EPS_L2    1e-12f

typedef __attribute__((ext_vector_type(16))) _Float16 v16h;
typedef __attribute__((ext_vector_type(8)))  float    v8f;

// ---------------------------------------------------------------
// f32 -> f16 elementwise cast
// ---------------------------------------------------------------
__global__ void cast_f32_f16(const float* __restrict__ in, _Float16* __restrict__ out, long n) {
    long i = (long)blockIdx.x * blockDim.x + threadIdx.x;
    if (i < n) out[i] = (_Float16)in[i];
}

// extract first TT rows of each batch of (B,81,E) ehs, cast to f16 -> (TTP,E),
// rows [B*TT, TTP) zero-filled so the GEMM needs no edge guards.
__global__ void cast_text_f16(const float* __restrict__ ehs, _Float16* __restrict__ out) {
    long i = (long)blockIdx.x * blockDim.x + threadIdx.x;
    long n = (long)TTP * EE;
    if (i >= n) return;
    int e = (int)(i % EE);
    int r = (int)(i / EE);          // padded row index
    if (r < BB * TT) {
        int t = r % TT;
        int b = r / TT;
        out[i] = (_Float16)ehs[((long)(b * T81 + t)) * EE + e];
    } else {
        out[i] = (_Float16)0.0f;
    }
}

// cast + transpose weight W (K x N, row-major f32) -> Wt (N x K, row-major f16)
__global__ void cast_transpose_f16(const float* __restrict__ W, _Float16* __restrict__ Wt,
                                   int K, int N) {
    long i = (long)blockIdx.x * blockDim.x + threadIdx.x;
    long n = (long)K * N;
    if (i >= n) return;
    int kk = (int)(i % K);
    int nn = (int)(i / K);
    Wt[(long)nn * K + kk] = (_Float16)W[(long)kk * N + nn];
}

// ---------------------------------------------------------------
// f16 WMMA GEMM: C[M,N] (f32) = A[M,K] (f16 row-major) * Bt[N,K]^T (f16 col-major-as-rows)
//                (+ bias[N]) (+ residual[M,N])
// One wave32 per 32x32 macro-tile: 2 A-frags x 2 B-frags -> 4 independent
// v_wmma_f32_16x16x32_f16 per K-step (operand reuse ~16 FLOP/B from L2).
// Requires: M % 32 == 0, N % 32 == 0, K % 32 == 0 (all callers padded).
// ---------------------------------------------------------------
__global__ __launch_bounds__(32)
void gemm_f16_wmma(const _Float16* __restrict__ A,
                   const _Float16* __restrict__ Bt,
                   float* __restrict__ C,
                   const float* __restrict__ bias,
                   const float* __restrict__ residual,
                   int M, int N, int K) {
    const int tiles_n2 = N >> 5;                 // 32-wide macro tiles in N
    const int tm   = (blockIdx.x / tiles_n2) * 2;  // 16-tile indices
    const int tn   = (blockIdx.x % tiles_n2) * 2;
    const int lane = threadIdx.x;                // 0..31
    const int half = lane >> 4;                  // hi 16 lanes
    const int lm   = lane & 15;

    // ISA 16-bit A 16x32 layout: lane(16*half+m): v0..3 = K[8*half..+7], v4..7 = K[16+8*half..+7]
    const _Float16* __restrict__ ap0 = A + (long)(tm * 16 + lm) * K;
    const _Float16* __restrict__ ap1 = ap0 + (long)16 * K;
    // ISA 16-bit B 32x16 layout: lane(16*half+n): v0..3 = K[8*half..+7], v4..7 = K[16+8*half..+7]
    const _Float16* __restrict__ bp0 = Bt + (long)(tn * 16 + lm) * K;
    const _Float16* __restrict__ bp1 = bp0 + (long)16 * K;

    v8f acc00 = {0.f,0.f,0.f,0.f,0.f,0.f,0.f,0.f};
    v8f acc01 = acc00, acc10 = acc00, acc11 = acc00;

    union Frag { v16h v; uint4 q[2]; };

    const int aoff = 8 * half;       // elements
    const int boff = 8 * half;

    #pragma unroll 2
    for (int k = 0; k < K; k += 32) {
        Frag a0, a1, b0, b1;
        a0.q[0] = *(const uint4*)(ap0 + k + aoff);
        a0.q[1] = *(const uint4*)(ap0 + k + 16 + aoff);
        a1.q[0] = *(const uint4*)(ap1 + k + aoff);
        a1.q[1] = *(const uint4*)(ap1 + k + 16 + aoff);
        b0.q[0] = *(const uint4*)(bp0 + k + boff);
        b0.q[1] = *(const uint4*)(bp0 + k + 16 + boff);
        b1.q[0] = *(const uint4*)(bp1 + k + boff);
        b1.q[1] = *(const uint4*)(bp1 + k + 16 + boff);

        // stream the next K-tiles toward the caches (global_prefetch_b8)
        __builtin_prefetch((const void*)(ap0 + k + 64), 0, 1);
        __builtin_prefetch((const void*)(bp0 + k + 64), 0, 1);

        acc00 = __builtin_amdgcn_wmma_f32_16x16x32_f16(false, a0.v, false, b0.v, (short)0, acc00, false, false);
        acc01 = __builtin_amdgcn_wmma_f32_16x16x32_f16(false, a0.v, false, b1.v, (short)0, acc01, false, false);
        acc10 = __builtin_amdgcn_wmma_f32_16x16x32_f16(false, a1.v, false, b0.v, (short)0, acc10, false, false);
        acc11 = __builtin_amdgcn_wmma_f32_16x16x32_f16(false, a1.v, false, b1.v, (short)0, acc11, false, false);
    }

    // C/D layout: VGPR r -> row = 16*tile_m + r + 8*half, col = 16*tile_n + lm
    #pragma unroll
    for (int i = 0; i < 2; ++i) {
        const v8f* accs = i ? &acc10 : &acc00;   // [i][0], [i][1] pairs below
        #pragma unroll
        for (int j = 0; j < 2; ++j) {
            const v8f& acc = (i == 0) ? (j == 0 ? acc00 : acc01)
                                      : (j == 0 ? acc10 : acc11);
            const int ccol = (tn + j) * 16 + lm;
            #pragma unroll
            for (int r = 0; r < 8; ++r) {
                const int crow = (tm + i) * 16 + r + 8 * half;
                long idx = (long)crow * N + ccol;
                float v = acc[r];
                if (bias)     v += bias[ccol];
                if (residual) v += residual[idx];
                C[idx] = v;
            }
        }
        (void)accs;
    }
}

// ---------------------------------------------------------------
// tiny scalar f32 kernels (each < 0.1 GFLOP): ip projections, normalization,
// Gram matrix M = X X^T, G = X ip_n^T
// ---------------------------------------------------------------
__global__ void ip_proj_kernel(const float* __restrict__ ehs,
                               const float* __restrict__ Wk,
                               const float* __restrict__ Wv,
                               float* __restrict__ k_ip, float* __restrict__ v_ip) {
    long i = (long)blockIdx.x * blockDim.x + threadIdx.x;
    long n = (long)BB * NTOK * CC;
    if (i >= n) return;
    int c  = (int)(i % CC);
    int nn = (int)((i / CC) % NTOK);
    int b  = (int)(i / ((long)CC * NTOK));
    const float* x = ehs + ((long)(b * T81 + TT + nn)) * EE;
    float ak = 0.f, av = 0.f;
    for (int e = 0; e < EE; ++e) {
        float xe = x[e];
        ak += xe * Wk[(long)e * CC + c];
        av += xe * Wv[(long)e * CC + c];
    }
    k_ip[i] = ak;
    v_ip[i] = av;
}

__global__ void ip_normalize_kernel(const float* __restrict__ ehs, float* __restrict__ ip_n) {
    // one block per ip row (B*NTOK blocks, 256 threads)
    int row = blockIdx.x;              // 0..7
    int nn = row % NTOK, b = row / NTOK;
    const float* x = ehs + ((long)(b * T81 + TT + nn)) * EE;
    __shared__ float red[256];
    float acc = 0.f;
    for (int e = threadIdx.x; e < EE; e += 256) { float v = x[e]; acc += v * v; }
    red[threadIdx.x] = acc;
    __syncthreads();
    for (int off = 128; off > 0; off >>= 1) {
        if (threadIdx.x < off) red[threadIdx.x] += red[threadIdx.x + off];
        __syncthreads();
    }
    float inv = 1.f / fmaxf(sqrtf(red[0]), EPS_L2);
    float* o = ip_n + (long)row * EE;
    for (int e = threadIdx.x; e < EE; e += 256) o[e] = x[e] * inv;
}

__global__ void gram_kernel(const float* __restrict__ ehs, float* __restrict__ Mg) {
    long i = (long)blockIdx.x * blockDim.x + threadIdx.x;
    long n = (long)BB * TT * TT;
    if (i >= n) return;
    int jj = (int)(i % TT);
    int ii = (int)((i / TT) % TT);
    int b  = (int)(i / ((long)TT * TT));
    const float* xi = ehs + ((long)(b * T81 + ii)) * EE;
    const float* xj = ehs + ((long)(b * T81 + jj)) * EE;
    float acc = 0.f;
    for (int e = 0; e < EE; ++e) acc += xi[e] * xj[e];
    Mg[i] = acc;
}

__global__ void g_kernel(const float* __restrict__ ehs, const float* __restrict__ ip_n,
                         float* __restrict__ G) {
    long i = (long)blockIdx.x * blockDim.x + threadIdx.x;
    long n = (long)BB * TT * NTOK;
    if (i >= n) return;
    int nn = (int)(i % NTOK);
    int t  = (int)((i / NTOK) % TT);
    int b  = (int)(i / ((long)NTOK * TT));
    const float* x = ehs + ((long)(b * T81 + t)) * EE;
    const float* y = ip_n + ((long)(b * NTOK + nn)) * EE;
    float acc = 0.f;
    for (int e = 0; e < EE; ++e) acc += x[e] * y[e];
    G[i] = acc;
}

// ---------------------------------------------------------------
// fused attention: per (b,h,s) row — text softmax, text_out,
// sem via Gram trick, 4-token fused IP softmax, ip_out
// grid = B*H*S blocks of 128 threads
// ---------------------------------------------------------------
__global__ __launch_bounds__(128)
void attn_kernel(const float* __restrict__ q,
                 const float* __restrict__ k_t, const float* __restrict__ v_t,
                 const float* __restrict__ k_ip, const float* __restrict__ v_ip,
                 const float* __restrict__ G,   const float* __restrict__ Mg,
                 const float* __restrict__ mask,
                 float* __restrict__ attn_out) {
    const int bid = blockIdx.x;
    const int s = bid % SS;
    const int h = (bid / SS) % HH;
    const int b = bid / (SS * HH);
    const int tid = threadIdx.x;
    const float scale = rsqrtf((float)DH);

    __shared__ float qv[DH];
    __shared__ float p[TT];
    __shared__ float rr[TT];
    __shared__ float sc4[NTOK], sg4[NTOK], spn[NTOK];
    __shared__ float s_invn;

    const float* qp = q + ((long)(b * SS + s)) * CC + h * DH;
    if (tid < DH) qv[tid] = qp[tid];
    __syncthreads();

    // text scores (k_t rows indexed by padded buffer: row = b*TT + t, valid t < TT)
    if (tid < TT) {
        const float* kp = k_t + ((long)(b * TT + tid)) * CC + h * DH;
        float acc = 0.f;
        #pragma unroll 8
        for (int d = 0; d < DH; ++d) acc += qv[d] * kp[d];
        p[tid] = acc * scale;
    }
    __syncthreads();

    // serial softmax over 77 (tiny)
    if (tid == 0) {
        float m = -3.4e38f;
        for (int t = 0; t < TT; ++t) m = fmaxf(m, p[t]);
        float sum = 0.f;
        for (int t = 0; t < TT; ++t) { float e = __expf(p[t] - m); p[t] = e; sum += e; }
        float inv = 1.f / sum;
        for (int t = 0; t < TT; ++t) p[t] *= inv;
    }
    __syncthreads();

    // text_out (regs), Gram quadratic-form partials, raw ip scores + p.G
    float to = 0.f;
    if (tid < DH) {
        for (int t = 0; t < TT; ++t)
            to += p[t] * v_t[((long)(b * TT + t)) * CC + h * DH + tid];
    }
    if (tid < TT) {
        const float* mrow = Mg + ((long)(b * TT + tid)) * TT;
        float acc = 0.f;
        for (int j = 0; j < TT; ++j) acc += mrow[j] * p[j];
        rr[tid] = acc * p[tid];
    }
    if (tid < NTOK) {
        const float* kp = k_ip + ((long)(b * NTOK + tid)) * CC + h * DH;
        float acc = 0.f;
        for (int d = 0; d < DH; ++d) acc += qv[d] * kp[d];
        float g = 0.f;
        for (int t = 0; t < TT; ++t) g += p[t] * G[((long)(b * TT + t)) * NTOK + tid];
        sc4[tid] = acc * scale;
        sg4[tid] = g;
    }
    __syncthreads();

    if (tid == 0) {
        float qq = 0.f;
        for (int t = 0; t < TT; ++t) qq += rr[t];
        s_invn = 1.f / fmaxf(sqrtf(fmaxf(qq, 0.f)), EPS_L2);
        // fused 4-way softmax * spatial mask
        float f[NTOK]; float m = -3.4e38f;
        for (int n = 0; n < NTOK; ++n) {
            f[n] = sc4[n] + SEM_SCALE * (sg4[n] * s_invn);
            m = fmaxf(m, f[n]);
        }
        float sum = 0.f;
        for (int n = 0; n < NTOK; ++n) { float e = __expf(f[n] - m); f[n] = e; sum += e; }
        float sc = mask[s] / sum;
        for (int n = 0; n < NTOK; ++n) spn[n] = f[n] * sc;
    }
    __syncthreads();

    if (tid < DH) {
        float ip = 0.f;
        #pragma unroll
        for (int n = 0; n < NTOK; ++n)
            ip += spn[n] * v_ip[((long)(b * NTOK + n)) * CC + h * DH + tid];
        attn_out[((long)(b * SS + s)) * CC + h * DH + tid] = to + IP_SCALE * ip;
    }
}

// ---------------------------------------------------------------
// launch
// ---------------------------------------------------------------
static inline size_t alignup(size_t x) { return (x + 255) & ~(size_t)255; }

extern "C" void kernel_launch(void* const* d_in, const int* in_sizes, int n_in,
                              void* d_out, int out_size, void* d_ws, size_t ws_size,
                              hipStream_t stream) {
    const float* hs     = (const float*)d_in[0];   // (B,S,C)
    const float* ehs    = (const float*)d_in[1];   // (B,81,E)
    const float* smask  = (const float*)d_in[2];   // (S,)
    const float* W_q    = (const float*)d_in[3];   // (C,C)
    const float* W_k    = (const float*)d_in[4];   // (E,C)
    const float* W_v    = (const float*)d_in[5];   // (E,C)
    const float* W_k_ip = (const float*)d_in[6];   // (E,C)
    const float* W_v_ip = (const float*)d_in[7];   // (E,C)
    const float* W_out  = (const float*)d_in[8];   // (C,C)
    const float* b_out  = (const float*)d_in[9];   // (C,)
    float* out = (float*)d_out;                    // (B,S,C)

    // ---- workspace carve-up ----
    char* w = (char*)d_ws;
    size_t off = 0;
    auto take = [&](size_t bytes) { char* p = w + off; off += alignup(bytes); return p; };

    _Float16* hs_f16   = (_Float16*)take((size_t)BB * SS * CC * 2);
    _Float16* text_f16 = (_Float16*)take((size_t)TTP * EE * 2);       // padded rows, tail zeroed
    _Float16* Wq_t     = (_Float16*)take((size_t)CC * CC * 2);
    _Float16* Wk_t     = (_Float16*)take((size_t)EE * CC * 2);
    _Float16* Wv_t     = (_Float16*)take((size_t)EE * CC * 2);
    _Float16* Wo_t     = (_Float16*)take((size_t)CC * CC * 2);
    float*    qbuf     = (float*)take((size_t)BB * SS * CC * 4);
    float*    k_t      = (float*)take((size_t)TTP * CC * 4);          // padded rows
    float*    v_t      = (float*)take((size_t)TTP * CC * 4);          // padded rows
    float*    k_ip     = (float*)take((size_t)BB * NTOK * CC * 4);
    float*    v_ip     = (float*)take((size_t)BB * NTOK * CC * 4);
    float*    ip_n     = (float*)take((size_t)BB * NTOK * EE * 4);
    float*    Gbuf     = (float*)take((size_t)BB * TT * NTOK * 4);
    float*    Mg       = (float*)take((size_t)BB * TT * TT * 4);
    float*    attn     = (float*)take((size_t)BB * SS * CC * 4);
    _Float16* attn_f16 = (_Float16*)take((size_t)BB * SS * CC * 2);
    (void)ws_size; (void)n_in; (void)in_sizes; (void)out_size;

    const int TPB = 256;
    auto blks = [](long n, int t) { return (int)((n + t - 1) / t); };

    // 1) casts
    cast_f32_f16<<<blks((long)BB * SS * CC, TPB), TPB, 0, stream>>>(hs, hs_f16, (long)BB * SS * CC);
    cast_text_f16<<<blks((long)TTP * EE, TPB), TPB, 0, stream>>>(ehs, text_f16);
    cast_transpose_f16<<<blks((long)CC * CC, TPB), TPB, 0, stream>>>(W_q,   Wq_t, CC, CC);
    cast_transpose_f16<<<blks((long)EE * CC, TPB), TPB, 0, stream>>>(W_k,   Wk_t, EE, CC);
    cast_transpose_f16<<<blks((long)EE * CC, TPB), TPB, 0, stream>>>(W_v,   Wv_t, EE, CC);
    cast_transpose_f16<<<blks((long)CC * CC, TPB), TPB, 0, stream>>>(W_out, Wo_t, CC, CC);

    // 2) big WMMA GEMMs (all dims multiples of 32 after padding)
    {
        int M = BB * SS, N = CC, K = CC;            // q projection
        gemm_f16_wmma<<<(M / 32) * (N / 32), 32, 0, stream>>>(
            hs_f16, Wq_t, qbuf, nullptr, nullptr, M, N, K);
    }
    {
        int M = TTP, N = CC, K = EE;                // k_t / v_t projections (padded M)
        int grid = (M / 32) * (N / 32);
        gemm_f16_wmma<<<grid, 32, 0, stream>>>(text_f16, Wk_t, k_t, nullptr, nullptr, M, N, K);
        gemm_f16_wmma<<<grid, 32, 0, stream>>>(text_f16, Wv_t, v_t, nullptr, nullptr, M, N, K);
    }

    // 3) small fp32 helpers
    ip_proj_kernel<<<blks((long)BB * NTOK * CC, TPB), TPB, 0, stream>>>(ehs, W_k_ip, W_v_ip, k_ip, v_ip);
    ip_normalize_kernel<<<BB * NTOK, 256, 0, stream>>>(ehs, ip_n);
    gram_kernel<<<blks((long)BB * TT * TT, TPB), TPB, 0, stream>>>(ehs, Mg);
    g_kernel<<<blks((long)BB * TT * NTOK, TPB), TPB, 0, stream>>>(ehs, ip_n, Gbuf);

    // 4) fused attention (text + sem-gated IP), fp32
    attn_kernel<<<BB * HH * SS, 128, 0, stream>>>(qbuf, k_t, v_t, k_ip, v_ip, Gbuf, Mg, smask, attn);

    // 5) output projection + bias + residual (WMMA)
    cast_f32_f16<<<blks((long)BB * SS * CC, TPB), TPB, 0, stream>>>(attn, attn_f16, (long)BB * SS * CC);
    {
        int M = BB * SS, N = CC, K = CC;
        gemm_f16_wmma<<<(M / 32) * (N / 32), 32, 0, stream>>>(
            attn_f16, Wo_t, out, b_out, hs, M, N, K);
    }
}
